// RoformerAttention_76012331204805
// MI455X (gfx1250) — compile-verified, matrix operand order checked by reading, not run on hardware
//
#include <hip/hip_runtime.h>
#include <hip/hip_bf16.h>

// ---------------------------------------------------------------------------
// CDNA5 (gfx1250) Roformer attention.
// bf16 WMMA everywhere; operands pre-packed to bf16; LDS tiles staged with
// GLOBAL_LOAD_ASYNC_TO_LDS_B128, double-buffered via counted S_WAIT_ASYNCCNT.
// ---------------------------------------------------------------------------

typedef __attribute__((ext_vector_type(16))) __bf16 v16bf;
typedef __attribute__((ext_vector_type(8)))  float  v8f;

union Frag16 { v16bf bf; unsigned int u[8]; };
union AccF   { v8f v; float f[8]; };

__device__ __forceinline__ unsigned short f2bf(float x) {
  union { float f; unsigned int u; } c; c.f = x;
  unsigned int u = c.u;
  u += 0x7FFFu + ((u >> 16) & 1u);   // round-to-nearest-even
  return (unsigned short)(u >> 16);
}
__device__ __forceinline__ float bf2f(unsigned short h) {
  union { unsigned int u; float f; } c; c.u = ((unsigned int)h) << 16;
  return c.f;
}
// K-pair (ushort) index inside a 16-bit A/B fragment: VGPR v, lane-half hf.
__device__ __forceinline__ int kk_of(int v, int hf) {
  return ((v >> 2) << 4) | (hf << 3) | ((v & 3) << 1);
}
__device__ __forceinline__ v8f wmma_bf16(const Frag16& a, const Frag16& b, v8f c) {
  return __builtin_amdgcn_wmma_f32_16x16x32_bf16(false, a.bf, false, b.bf,
                                                 (short)0, c, false, false);
}
// 16B global -> LDS async copy (ASYNCcnt tracked, in-order completion).
__device__ __forceinline__ void async_b128(void* lds, const void* g) {
  unsigned int off = (unsigned int)(size_t)lds;            // low 32b = LDS offset
  unsigned long long ga = (unsigned long long)(size_t)g;
  asm volatile("global_load_async_to_lds_b128 %0, %1, off"
               :: "v"(off), "v"(ga) : "memory");
}
__device__ __forceinline__ void wait_async0() {
  asm volatile("s_wait_asynccnt 0x0" ::: "memory");
}
__device__ __forceinline__ void wait_async4() {
  asm volatile("s_wait_asynccnt 0x4" ::: "memory");
}

// ---------------------------------------------------------------------------
// One-shot packers.
// ---------------------------------------------------------------------------
__global__ __launch_bounds__(256)
void cvt_bf16(const float* __restrict__ in, unsigned short* __restrict__ out, long n) {
  long i = (long)blockIdx.x * blockDim.x + threadIdx.x;
  if (i < n) out[i] = f2bf(in[i]);
}
// in[K][N] f32 -> out[N][K] bf16
__global__ __launch_bounds__(256)
void cvt_bf16_t(const float* __restrict__ in, unsigned short* __restrict__ out,
                int K, int N) {
  long i = (long)blockIdx.x * blockDim.x + threadIdx.x;
  if (i >= (long)K * N) return;
  int k = (int)(i / N), n = (int)(i % N);
  out[(size_t)n * K + k] = f2bf(in[i]);
}

// ---------------------------------------------------------------------------
// GEMM: C = A[M,K](bf16) @ Bt[N,K](bf16, row=n) + bias, f32 accumulate.
// Block 256 threads (8 waves), tile 128x128, K-slab 32, double-buffered
// async LDS staging.
// MODE 0: C32[M,N] f32; MODE 1: C16[M,N] bf16; MODE 2: C16 as [B,H,64,S] bf16.
// ---------------------------------------------------------------------------
template <int MODE>
__global__ __launch_bounds__(256)
void gemm_bf16(const unsigned short* __restrict__ A,
               const unsigned short* __restrict__ Bt,
               const float* __restrict__ bias,
               float* __restrict__ C32, unsigned short* __restrict__ C16,
               int M, int N, int K, int S, int H)
{
  // padded rows: 32 bf16 data + 8 pad = 40 ushorts = 80B (16B aligned)
  __shared__ __align__(16) unsigned short As[2][128 * 40];
  __shared__ __align__(16) unsigned short Bs[2][128 * 40];

  const int tid  = threadIdx.x;
  const int lane = tid & 31;
  const int wave = tid >> 5;
  const int wr   = wave >> 1;            // 0..3
  const int wc   = wave & 1;             // 0..1
  const int hf   = lane >> 4;
  const int cn   = lane & 15;

  const int rowBase = blockIdx.y * 128;
  const int colBase = blockIdx.x * 128;

  AccF acc[2][4];
#pragma unroll
  for (int i = 0; i < 2; ++i)
#pragma unroll
    for (int j = 0; j < 4; ++j) acc[i][j].v = {};

  auto stage = [&](int buf, int k0) {
#pragma unroll
    for (int i = 0; i < 2; ++i) {
      const int c   = tid + 256 * i;
      const int row = c >> 2, seg = c & 3;
      async_b128(&As[buf][row * 40 + seg * 8],
                 A  + (size_t)(rowBase + row) * K + k0 + seg * 8);
      async_b128(&Bs[buf][row * 40 + seg * 8],
                 Bt + (size_t)(colBase + row) * K + k0 + seg * 8);
    }
  };

  const int nIter = K >> 5;
  stage(0, 0);
  for (int it = 0; it < nIter; ++it) {
    const int buf = it & 1;
    if (it + 1 < nIter) { stage(buf ^ 1, (it + 1) << 5); wait_async4(); }
    else                { wait_async0(); }
    __syncthreads();

    Frag16 af[2];
#pragma unroll
    for (int i = 0; i < 2; ++i) {
      const int m = wr * 32 + i * 16 + cn;
#pragma unroll
      for (int v = 0; v < 8; ++v)
        af[i].u[v] = *(const unsigned int*)&As[buf][m * 40 + kk_of(v, hf)];
    }
    Frag16 bfr[4];
#pragma unroll
    for (int j = 0; j < 4; ++j) {
      const int n = wc * 64 + j * 16 + cn;
#pragma unroll
      for (int v = 0; v < 8; ++v)
        bfr[j].u[v] = *(const unsigned int*)&Bs[buf][n * 40 + kk_of(v, hf)];
    }
#pragma unroll
    for (int i = 0; i < 2; ++i)
#pragma unroll
      for (int j = 0; j < 4; ++j)
        acc[i][j].v = wmma_bf16(af[i], bfr[j], acc[i][j].v);

    __syncthreads();
  }

  // Epilogue (all index math block/tile-hoisted; MODE is compile-time).
  if constexpr (MODE == 2) {
    const int bb    = rowBase / S;               // 128-row tiles never straddle
    const int srow0 = rowBase - bb * S + wr * 32;
#pragma unroll
    for (int j = 0; j < 4; ++j) {
      const int col = colBase + wc * 64 + j * 16 + cn;
      const float bv = bias[col];
      const int h = col >> 6, d = col & 63;
      unsigned short* base = C16 + (((size_t)bb * H + h) * 64 + d) * S + srow0;
#pragma unroll
      for (int i = 0; i < 2; ++i)
#pragma unroll
        for (int r = 0; r < 8; ++r)
          base[i * 16 + r + 8 * hf] = f2bf(acc[i][j].f[r] + bv);
    }
  } else {
#pragma unroll
    for (int j = 0; j < 4; ++j) {
      const int col = colBase + wc * 64 + j * 16 + cn;
      const float bv = bias[col];
      const size_t rb = (size_t)(rowBase + wr * 32) * N + col;
#pragma unroll
      for (int i = 0; i < 2; ++i) {
#pragma unroll
        for (int r = 0; r < 8; ++r) {
          const size_t off = rb + (size_t)(i * 16 + r + 8 * hf) * N;
          const float val = acc[i][j].f[r] + bv;
          if constexpr (MODE == 0) C32[off] = val;
          else                     C16[off] = f2bf(val);
        }
      }
    }
  }
}

// ---------------------------------------------------------------------------
// In-place RoPE on bf16 Q,K ([B,S,D]); full-head-dim frequency table.
// ---------------------------------------------------------------------------
__global__ __launch_bounds__(256)
void rope_bf16(unsigned short* __restrict__ Q, unsigned short* __restrict__ K,
               int S, int H, int hd, long total)
{
  long idx = (long)blockIdx.x * blockDim.x + threadIdx.x;
  if (idx >= total) return;
  const int jh = hd >> 1;
  int j = (int)(idx % jh); long t = idx / jh;
  int h = (int)(t % H);    t /= H;
  int s = (int)(t % S);
  int b = (int)(t / S);
  const int D = H * hd;
  const size_t base = ((size_t)(b * S + s)) * D + h * hd;

  const float LOG1E4 = 9.210340371976184f;
  float inv1 = __expf(-((float)j        / (float)hd) * LOG1E4);
  float inv2 = __expf(-((float)(j + jh) / (float)hd) * LOG1E4);
  float a1 = (float)s * inv1, a2 = (float)s * inv2;
  float c1 = __cosf(a1), s1 = __sinf(a1);
  float c2 = __cosf(a2), s2 = __sinf(a2);

  float q1 = bf2f(Q[base + j]), q2 = bf2f(Q[base + j + jh]);
  Q[base + j]      = f2bf(q1 * c1 - q2 * s1);
  Q[base + j + jh] = f2bf(q2 * c2 + q1 * s2);

  float k1 = bf2f(K[base + j]), k2 = bf2f(K[base + j + jh]);
  K[base + j]      = f2bf(k1 * c1 - k2 * s1);
  K[base + j + jh] = f2bf(k2 * c2 + k1 * s2);
}

// ---------------------------------------------------------------------------
// Flash attention. Grid (S/64, H, B); 128 threads = 4 waves x 16 q-rows.
// Kb: bf16 [B,S,D]; Vt: bf16 [B,H,64,S]; mask f32 [B,1,S,S]; out bf16 [B,S,D].
// Double-buffered async K/V tile staging.
// ---------------------------------------------------------------------------
__global__ __launch_bounds__(128)
void attention_kernel(const unsigned short* __restrict__ Qb,
                      const unsigned short* __restrict__ Kb,
                      const unsigned short* __restrict__ Vt,
                      const float* __restrict__ mask,
                      const float* __restrict__ cdrw_p,
                      unsigned short* __restrict__ O,
                      int H, int S, int hd)
{
  // Ks: 32 keys x 64 d  -> row stride 72 ushorts (144B, 16B aligned)
  // Vs: 64 d    x 32 key-> row stride 40 ushorts (80B, 16B aligned)
  __shared__ __align__(16) unsigned short Ks[2][32 * 72];
  __shared__ __align__(16) unsigned short Vs[2][64 * 40];
  __shared__ __align__(16) unsigned short Ps[4][16][34];

  const int b = blockIdx.z, h = blockIdx.y;
  const int tid = threadIdx.x, lane = tid & 31, wave = tid >> 5;
  const int hf = lane >> 4, cn = lane & 15;
  const int qbase = blockIdx.x * 64 + wave * 16;
  const int D = H * hd;
  const float cdrw = cdrw_p[0];

  // Q fragments (16 rows, GEMM-K 64 = 2 chunks): direct packed-bf16 loads.
  Frag16 qf[2];
  {
    const unsigned short* qp = Qb + ((size_t)(b * S + qbase + cn)) * D + h * hd;
#pragma unroll
    for (int c = 0; c < 2; ++c)
#pragma unroll
      for (int v = 0; v < 8; ++v)
        qf[c].u[v] = *(const unsigned int*)(qp + c * 32 + kk_of(v, hf));
  }

  float mrow[8], lrow[8];
#pragma unroll
  for (int r = 0; r < 8; ++r) { mrow[r] = -1e30f; lrow[r] = 0.0f; }
  AccF o[4];
#pragma unroll
  for (int t = 0; t < 4; ++t) o[t].v = {};

  const float* maskB = mask + (size_t)b * S * S;
  const unsigned short* KbH = Kb + (size_t)b * S * D + h * hd;
  const unsigned short* VtH = Vt + ((size_t)b * H + h) * 64 * S;

  auto stage = [&](int bufi, int kb) {
#pragma unroll
    for (int i = 0; i < 2; ++i) {
      const int c = tid + 128 * i;
      {
        const int row = c >> 3, seg = c & 7;
        async_b128(&Ks[bufi][row * 72 + seg * 8],
                   KbH + (size_t)(kb + row) * D + seg * 8);
      }
      {
        const int row = c >> 2, seg = c & 3;
        async_b128(&Vs[bufi][row * 40 + seg * 8],
                   VtH + (size_t)row * S + kb + seg * 8);
      }
    }
  };

  const int nIter = S >> 5;
  stage(0, 0);
  for (int it = 0; it < nIter; ++it) {
    const int buf = it & 1;
    const int kb  = it << 5;
    if (it + 1 < nIter) { stage(buf ^ 1, (it + 1) << 5); wait_async4(); }
    else                { wait_async0(); }
    __syncthreads();

    // Scores: 16 x 32 = two 16x16 accumulators, GEMM-K = 64.
    AccF sc[2];
#pragma unroll
    for (int jt = 0; jt < 2; ++jt) {
      sc[jt].v = {};
      const int keyc = jt * 16 + cn;
#pragma unroll
      for (int c = 0; c < 2; ++c) {
        Frag16 bk;
#pragma unroll
        for (int v = 0; v < 8; ++v)
          bk.u[v] = *(const unsigned int*)&Ks[buf][keyc * 72 + c * 32 + kk_of(v, hf)];
        sc[jt].v = wmma_bf16(qf[c], bk, sc[jt].v);
      }
    }

    // Scale + multiplicative mask.
#pragma unroll
    for (int jt = 0; jt < 2; ++jt) {
      const size_t kcol = (size_t)(kb + jt * 16 + cn);
#pragma unroll
      for (int r = 0; r < 8; ++r) {
        const size_t qrow = (size_t)(qbase + r + 8 * hf);
        float mv = maskB[qrow * S + kcol];
        sc[jt].f[r] = sc[jt].f[r] * 0.125f * (1.0f + mv * cdrw);
      }
    }

    // Online softmax (16-lane xor reductions within each half).
    float scl[8];
#pragma unroll
    for (int r = 0; r < 8; ++r) {
      float v = fmaxf(sc[0].f[r], sc[1].f[r]);
      v = fmaxf(v, __shfl_xor(v, 1, 32));
      v = fmaxf(v, __shfl_xor(v, 2, 32));
      v = fmaxf(v, __shfl_xor(v, 4, 32));
      v = fmaxf(v, __shfl_xor(v, 8, 32));
      float mn = fmaxf(mrow[r], v);
      scl[r] = __expf(mrow[r] - mn);
      mrow[r] = mn;
      lrow[r] *= scl[r];
    }
#pragma unroll
    for (int t = 0; t < 4; ++t)
#pragma unroll
      for (int r = 0; r < 8; ++r) o[t].f[r] *= scl[r];

#pragma unroll
    for (int jt = 0; jt < 2; ++jt)
#pragma unroll
      for (int r = 0; r < 8; ++r)
        sc[jt].f[r] = __expf(sc[jt].f[r] - mrow[r]);
#pragma unroll
    for (int r = 0; r < 8; ++r) {
      float v = sc[0].f[r] + sc[1].f[r];
      v += __shfl_xor(v, 1, 32);
      v += __shfl_xor(v, 2, 32);
      v += __shfl_xor(v, 4, 32);
      v += __shfl_xor(v, 8, 32);
      lrow[r] += v;
    }

    // Relayout P (C-frag) -> A-frag via per-wave LDS.
#pragma unroll
    for (int jt = 0; jt < 2; ++jt)
#pragma unroll
      for (int r = 0; r < 8; ++r)
        Ps[wave][r + 8 * hf][jt * 16 + cn] = f2bf(sc[jt].f[r]);
    asm volatile("s_wait_dscnt 0x0" ::: "memory");

    Frag16 pf;
#pragma unroll
    for (int v = 0; v < 8; ++v)
      pf.u[v] = *(const unsigned int*)&Ps[wave][cn][kk_of(v, hf)];

    // O += P @ V (GEMM-K = 32, 4 d-tiles); key-pairs contiguous in Vs rows.
#pragma unroll
    for (int t = 0; t < 4; ++t) {
      Frag16 bv;
#pragma unroll
      for (int v = 0; v < 8; ++v)
        bv.u[v] = *(const unsigned int*)&Vs[buf][(t * 16 + cn) * 40 + kk_of(v, hf)];
      o[t].v = wmma_bf16(pf, bv, o[t].v);
    }
    __syncthreads();
  }

  // Normalize; write bf16 [B,S,D] for the final projection GEMM.
#pragma unroll
  for (int r = 0; r < 8; ++r) {
    const float inv = 1.0f / lrow[r];
    const size_t qrow = (size_t)(qbase + r + 8 * hf);
#pragma unroll
    for (int t = 0; t < 4; ++t) {
      const int dcol = t * 16 + cn;
      O[((size_t)b * S + qrow) * D + h * hd + dcol] = f2bf(o[t].f[r] * inv);
    }
  }
}

// ---------------------------------------------------------------------------
// Host-side launch
// ---------------------------------------------------------------------------
extern "C" void kernel_launch(void* const* d_in, const int* in_sizes, int n_in,
                              void* d_out, int out_size, void* d_ws, size_t ws_size,
                              hipStream_t stream) {
  const float* x    = (const float*)d_in[0];
  const float* msk  = (const float*)d_in[1];
  const float* Wq   = (const float*)d_in[2];
  const float* bq   = (const float*)d_in[3];
  const float* Wk   = (const float*)d_in[4];
  const float* bk   = (const float*)d_in[5];
  const float* Wv   = (const float*)d_in[6];
  const float* bv   = (const float*)d_in[7];
  const float* Wo   = (const float*)d_in[8];
  const float* bo   = (const float*)d_in[9];
  const float* cdrw = (const float*)d_in[10];
  float* out = (float*)d_out;

  const int  D  = in_sizes[3];                 // 1024
  const long BS = (long)in_sizes[0] / D;       // 4096
  const long S  = (long)in_sizes[1] / BS;      // 2048
  const int  Bn = (int)(BS / S);               // 2
  const int  H  = 16;
  const int  hd = D / H;                       // 64

  const size_t PLANE = (size_t)BS * D;         // elements of a [B,S,D] tensor
  const size_t WSZ   = (size_t)D * D;

  unsigned short* xb  = (unsigned short*)d_ws;
  unsigned short* Wqt = xb  + PLANE;
  unsigned short* Wkt = Wqt + WSZ;
  unsigned short* Wvt = Wkt + WSZ;
  unsigned short* Wot = Wvt + WSZ;
  unsigned short* Qb  = Wot + WSZ;
  unsigned short* Kb  = Qb  + PLANE;
  unsigned short* Vt  = Kb  + PLANE;
  unsigned short* AOb = Vt  + PLANE;

  // Pack inputs to bf16 (weights transposed to [N,K]).
  cvt_bf16<<<(unsigned)((PLANE + 255) / 256), 256, 0, stream>>>(x, xb, (long)PLANE);
  unsigned wgrid = (unsigned)((WSZ + 255) / 256);
  cvt_bf16_t<<<wgrid, 256, 0, stream>>>(Wq, Wqt, D, D);
  cvt_bf16_t<<<wgrid, 256, 0, stream>>>(Wk, Wkt, D, D);
  cvt_bf16_t<<<wgrid, 256, 0, stream>>>(Wv, Wvt, D, D);
  cvt_bf16_t<<<wgrid, 256, 0, stream>>>(Wo, Wot, D, D);

  dim3 gGrid((unsigned)(D / 128), (unsigned)(BS / 128));
  // Q, K as bf16 [M,N]; V as bf16 [B,H,64,S].
  gemm_bf16<1><<<gGrid, 256, 0, stream>>>(xb, Wqt, bq, nullptr, Qb,
                                          (int)BS, D, D, (int)S, H);
  gemm_bf16<1><<<gGrid, 256, 0, stream>>>(xb, Wkt, bk, nullptr, Kb,
                                          (int)BS, D, D, (int)S, H);
  gemm_bf16<2><<<gGrid, 256, 0, stream>>>(xb, Wvt, bv, nullptr, Vt,
                                          (int)BS, D, D, (int)S, H);

  long total = (long)Bn * S * H * (hd / 2);
  rope_bf16<<<(unsigned)((total + 255) / 256), 256, 0, stream>>>(
      Qb, Kb, (int)S, H, hd, total);

  dim3 aGrid((unsigned)(S / 64), (unsigned)H, (unsigned)Bn);
  attention_kernel<<<aGrid, 128, 0, stream>>>(Qb, Kb, Vt, msk, cdrw, AOb,
                                              H, (int)S, hd);

  gemm_bf16<0><<<gGrid, 256, 0, stream>>>(AOb, Wot, bo, out, nullptr,
                                          (int)BS, D, D, (int)S, H);
}
